// MambaBlock_61246233641281
// MI455X (gfx1250) — compile-verified, matrix operand order checked by reading, not run on hardware
//
#include <hip/hip_runtime.h>
#include <hip/hip_bf16.h>

typedef __attribute__((ext_vector_type(16))) __bf16 v16bf;
typedef __attribute__((ext_vector_type(4)))  __bf16 v4bf;
typedef __attribute__((ext_vector_type(8)))  float  v8f;

#define DEV_INLINE __device__ __forceinline__

// Problem constants
static constexpr int BB  = 2;
static constexpr int LL  = 2048;
static constexpr int DD  = 1024;
static constexpr int DI_ = 2048;
static constexpr int NN  = 16;
static constexpr int RR  = 64;

// ---------------------------------------------------------------------------
// bf16 fragment load: 16 bf16 per lane for one 16x32 operand tile.
// ISA 7.12.2 (16-bit A 16x32): lanes 0-15 hold K = {kb+0..7, kb+16..23},
// lanes 16-31 hold K = {kb+8..15, kb+24..31}; AMD table element order.
// `p` points at base + row*ld + kb + half*8 (bf16 elements, 16B aligned).
// ---------------------------------------------------------------------------
DEV_INLINE v16bf ldfrag_bf16(const __bf16* __restrict__ p) {
  union { v16bf v; float4 f[2]; } r;
  r.f[0] = *(const float4*)(p);
  r.f[1] = *(const float4*)(p + 16);
  return r.v;
}

// f32 source fragment: load all raw data first, convert after (keeps loads
// independent so they can be clause-issued together).
DEV_INLINE v16bf ldfrag_f32(const float* __restrict__ p) {
  float4 f0 = ((const float4*)p)[0];
  float4 f1 = ((const float4*)p)[1];
  float4 f2 = ((const float4*)(p + 16))[0];
  float4 f3 = ((const float4*)(p + 16))[1];
  v16bf r;
  r[0]  = (__bf16)f0.x; r[1]  = (__bf16)f0.y; r[2]  = (__bf16)f0.z; r[3]  = (__bf16)f0.w;
  r[4]  = (__bf16)f1.x; r[5]  = (__bf16)f1.y; r[6]  = (__bf16)f1.z; r[7]  = (__bf16)f1.w;
  r[8]  = (__bf16)f2.x; r[9]  = (__bf16)f2.y; r[10] = (__bf16)f2.z; r[11] = (__bf16)f2.w;
  r[12] = (__bf16)f3.x; r[13] = (__bf16)f3.y; r[14] = (__bf16)f3.z; r[15] = (__bf16)f3.w;
  return r;
}

DEV_INLINE float softplus_f(float v) {
  return (v > 20.0f) ? v : log1pf(__expf(v));
}

// ---------------------------------------------------------------------------
// C[M,N] = act( A[M,K] @ W[N,K]^T + bias[N] ),  W in bf16.
// ABF: A operand is bf16 (true) or f32-with-inline-cvt (false).
// Each wave: one 16-row tile x NT 16-col tiles. tilesN % NT == 0 -> the K-loop
// body is completely branch-free (no exec masking near WMMA).
// All NT B-fragments are loaded before any WMMA so loads clause together.
// ---------------------------------------------------------------------------
template<int ACT, int NT, bool ABF>
__global__ __launch_bounds__(256)
void gemm_wmma_kernel(const void* __restrict__ Av, int lda,
                      const __bf16* __restrict__ W, int ldw,
                      const float* __restrict__ bias,
                      float* __restrict__ C, int ldc,
                      int M, int N, int K) {
  const int lane = threadIdx.x & 31;
  const int wid  = threadIdx.x >> 5;
  const int half = lane >> 4;
  const int lrow = lane & 15;

  const int tilesM = M >> 4;
  const int jobsN  = (N >> 4) / NT;      // exact by construction

  const int job = blockIdx.x * 8 + wid;  // wave-uniform
  if (job >= tilesM * jobsN) return;
  const int tm  = job / jobsN;
  const int tng = job % jobsN;

  v8f acc[NT];
#pragma unroll
  for (int j = 0; j < NT; ++j)
#pragma unroll
    for (int i = 0; i < 8; ++i) acc[j][i] = 0.0f;

  const __bf16* aRowB = (const __bf16*)Av + (size_t)(tm * 16 + lrow) * lda + half * 8;
  const float*  aRowF = (const float*) Av + (size_t)(tm * 16 + lrow) * lda + half * 8;
  const __bf16* bRow[NT];
#pragma unroll
  for (int j = 0; j < NT; ++j)
    bRow[j] = W + (size_t)((tng * NT + j) * 16 + lrow) * ldw + half * 8;

#pragma unroll 2
  for (int kb = 0; kb < K; kb += 32) {
    v16bf afrag;
    if (ABF) {
      // locality 3 -> WGP-scope prefetch (pulls into all cache levels);
      // speculative, OOB translation failures silently dropped (ISA 10.5).
      __builtin_prefetch(aRowB + kb + 32, 0, 3);
      afrag = ldfrag_bf16(aRowB + kb);
    } else {
      afrag = ldfrag_f32(aRowF + kb);
    }
    v16bf bfrag[NT];
#pragma unroll
    for (int j = 0; j < NT; ++j)
      bfrag[j] = ldfrag_bf16(bRow[j] + kb);
#pragma unroll
    for (int j = 0; j < NT; ++j)
      acc[j] = __builtin_amdgcn_wmma_f32_16x16x32_bf16(
          false, afrag, false, bfrag[j], (short)0, acc[j], false, false);
  }

  // C layout (ISA 7.12.2): VGPR r -> M = tm*16 + r + half*8, N = tn*16 + lrow
#pragma unroll
  for (int j = 0; j < NT; ++j) {
    const int col = (tng * NT + j) * 16 + lrow;
    const float bv = bias[col];
#pragma unroll
    for (int r = 0; r < 8; ++r) {
      const int row = tm * 16 + r + half * 8;
      float v = acc[j][r] + bv;
      if (ACT == 1) v = softplus_f(v);
      C[(size_t)row * ldc + col] = v;
    }
  }
}

// ---------------------------------------------------------------------------
// f32 -> bf16 bulk convert, 4 elements per thread.
// ---------------------------------------------------------------------------
__global__ __launch_bounds__(256)
void cvt4_f32_bf16_kernel(const float* __restrict__ in,
                          __bf16* __restrict__ out, int n4) {
  const int i = blockIdx.x * blockDim.x + threadIdx.x;
  if (i >= n4) return;
  float4 f = ((const float4*)in)[i];
  v4bf o;
  o[0] = (__bf16)f.x; o[1] = (__bf16)f.y; o[2] = (__bf16)f.z; o[3] = (__bf16)f.w;
  ((v4bf*)out)[i] = o;
}

// ---------------------------------------------------------------------------
// Depthwise conv1d (kernel 3, pad 1) + bias + SiLU. Writes u (f32, for the
// scan) and ub (bf16, for the WMMA GEMMs).
// ---------------------------------------------------------------------------
__global__ __launch_bounds__(256)
void conv_silu_kernel(const float* __restrict__ h,
                      const float* __restrict__ conv_w,
                      const float* __restrict__ conv_b,
                      float* __restrict__ u,
                      __bf16* __restrict__ ub) {
  const size_t idx   = (size_t)blockIdx.x * blockDim.x + threadIdx.x;
  const size_t total = (size_t)BB * LL * DI_;
  if (idx >= total) return;
  const int di = (int)(idx % DI_);
  const size_t bl = idx / DI_;
  const int l = (int)(bl % LL);
  const int b = (int)(bl / LL);

  const float w0 = conv_w[di * 3 + 0];
  const float w1 = conv_w[di * 3 + 1];
  const float w2 = conv_w[di * 3 + 2];
  const float* hb = h + (size_t)b * LL * DI_;

  float acc = conv_b[di];
  if (l > 0)      acc += hb[(size_t)(l - 1) * DI_ + di] * w0;
                  acc += hb[(size_t)(l    ) * DI_ + di] * w1;
  if (l < LL - 1) acc += hb[(size_t)(l + 1) * DI_ + di] * w2;

  const float s = acc / (1.0f + __expf(-acc));   // silu
  u[idx]  = s;
  ub[idx] = (__bf16)s;
}

// ---------------------------------------------------------------------------
// Fused selective scan. One thread per (b, di) channel; 16-state carry in
// registers; sequential over L. Output y written directly as bf16 (only the
// out-projection GEMM consumes it).
// ---------------------------------------------------------------------------
__global__ __launch_bounds__(256)
void scan_kernel(const float* __restrict__ delta,
                 const float* __restrict__ u,
                 const float* __restrict__ xdbl,   // (B, L, R+N); Bm at offset R
                 const float* __restrict__ A_log,
                 const float* __restrict__ D_param,
                 __bf16* __restrict__ yb) {
  const int ch = blockIdx.x * blockDim.x + threadIdx.x;
  if (ch >= BB * DI_) return;
  const int b  = ch / DI_;
  const int di = ch % DI_;

  float Arow[NN], carry[NN];
#pragma unroll
  for (int n = 0; n < NN; ++n) {
    Arow[n]  = -__expf(A_log[di * NN + n]);
    carry[n] = 0.0f;
  }
  const float Dp = D_param[di];

  const float* dptr = delta + (size_t)b * LL * DI_ + di;
  const float* uptr = u     + (size_t)b * LL * DI_ + di;
  const float* bptr = xdbl  + (size_t)b * LL * (RR + NN) + RR;
  __bf16*      yptr = yb    + (size_t)b * LL * DI_ + di;

  for (int l = 0; l < LL; ++l) {
    const float dl = dptr[(size_t)l * DI_];
    const float uu = uptr[(size_t)l * DI_];
    const float du = dl * uu;
    const float* Bl = bptr + (size_t)l * (RR + NN);
    float sum = 0.0f;
#pragma unroll
    for (int n = 0; n < NN; ++n) {
      const float dA = __expf(dl * Arow[n]);
      carry[n] = dA * carry[n] + du * Bl[n];
      sum += carry[n];
    }
    yptr[(size_t)l * DI_] = (__bf16)(sum + uu * Dp);
  }
}

// ---------------------------------------------------------------------------
// Host-side launchers
// ---------------------------------------------------------------------------
template<int ACT, int NT, bool ABF>
static void launch_gemm(const void* A, int lda, const __bf16* W, int ldw,
                        const float* bias, float* C, int ldc,
                        int M, int N, int K, hipStream_t stream) {
  const int jobs   = (M / 16) * ((N / 16) / NT);
  const int blocks = (jobs + 7) / 8;            // 8 waves per 256-thread block
  gemm_wmma_kernel<ACT, NT, ABF><<<blocks, 256, 0, stream>>>(
      A, lda, W, ldw, bias, C, ldc, M, N, K);
}

static void launch_cvt(const float* in, __bf16* out, size_t n, hipStream_t stream) {
  const int n4 = (int)(n / 4);
  cvt4_f32_bf16_kernel<<<(n4 + 255) / 256, 256, 0, stream>>>(in, out, n4);
}

extern "C" void kernel_launch(void* const* d_in, const int* in_sizes, int n_in,
                              void* d_out, int out_size, void* d_ws, size_t ws_size,
                              hipStream_t stream) {
  const float* x       = (const float*)d_in[0];   // (B,L,D)
  const float* in_w    = (const float*)d_in[1];   // (DI,D)
  const float* in_b    = (const float*)d_in[2];   // (DI)
  const float* conv_w  = (const float*)d_in[3];   // (DI,1,3)
  const float* conv_b  = (const float*)d_in[4];   // (DI)
  const float* xproj_w = (const float*)d_in[5];   // (R+N,DI)
  const float* xproj_b = (const float*)d_in[6];   // (R+N)
  const float* dt_w    = (const float*)d_in[7];   // (DI,R)
  const float* dt_b    = (const float*)d_in[8];   // (DI)
  const float* A_log   = (const float*)d_in[9];   // (DI,N)
  const float* D_param = (const float*)d_in[10];  // (DI)
  const float* out_w   = (const float*)d_in[11];  // (D,DI)
  const float* out_b   = (const float*)d_in[12];  // (D)
  float* out = (float*)d_out;                     // (B,L,D) f32

  // ---- workspace layout -----------------------------------------------
  const size_t nBLDI = (size_t)BB * LL * DI_;       // 8,388,608
  const size_t nBLD  = (size_t)BB * LL * DD;        // 4,194,304
  const size_t nXDBL = (size_t)BB * LL * (RR + NN); //   327,680

  float* h     = (float*)d_ws;            // also 'delta' (h dead after conv)
  float* u     = h + nBLDI;
  float* xdbl  = u + nBLDI;
  float* delta = h;

  __bf16* xb    = (__bf16*)(xdbl + nXDBL);
  __bf16* ub    = xb    + nBLD;
  __bf16* yb    = ub    + nBLDI;
  __bf16* inwb  = yb    + nBLDI;
  __bf16* xpwb  = inwb  + (size_t)DI_ * DD;
  __bf16* dtwb  = xpwb  + (size_t)(RR + NN) * DI_;
  __bf16* outwb = dtwb  + (size_t)DI_ * RR;

  const int M = BB * LL;   // 4096

  // ---- 0) one-time-per-call f32 -> bf16 conversions --------------------
  launch_cvt(x,       xb,    nBLD,                  stream);
  launch_cvt(in_w,    inwb,  (size_t)DI_ * DD,      stream);
  launch_cvt(xproj_w, xpwb,  (size_t)(RR + NN) * DI_, stream);
  launch_cvt(dt_w,    dtwb,  (size_t)DI_ * RR,      stream);
  launch_cvt(out_w,   outwb, (size_t)DD * DI_,      stream);

  // ---- 1) h = x @ in_w^T + in_b          (M x DI, K = D)   NT=4
  launch_gemm<0, 4, true>(xb, DD, inwb, DD, in_b, h, DI_, M, DI_, DD, stream);

  // ---- 2) u = silu(conv3(h) + conv_b)  (f32 + bf16 copies)
  conv_silu_kernel<<<(unsigned)((nBLDI + 255) / 256), 256, 0, stream>>>(
      h, conv_w, conv_b, u, ub);

  // ---- 3) x_dbl = u @ x_proj_w^T + b     (M x 80, K = DI)  NT=5
  launch_gemm<0, 5, true>(ub, DI_, xpwb, DI_, xproj_b, xdbl, RR + NN,
                          M, RR + NN, DI_, stream);

  // ---- 4) delta = softplus(dl @ dt_w^T + b)  (M x DI, K = R; A f32 stride 80)
  launch_gemm<1, 4, false>(xdbl, RR + NN, dtwb, RR, dt_b, delta, DI_,
                           M, DI_, RR, stream);

  // ---- 5) fused selective scan -> yb (bf16)
  scan_kernel<<<(BB * DI_ + 255) / 256, 256, 0, stream>>>(
      delta, u, xdbl, A_log, D_param, yb);

  // ---- 6) out = y @ out_w^T + out_b      (M x D, K = DI)   NT=4
  launch_gemm<0, 4, true>(yb, DI_, outwb, DI_, out_b, out, DD, M, DD, DI_, stream);
}
